// MixLoraSparseMoe_11845519802948
// MI455X (gfx1250) — matrix operand order repaired and checked by reading, not compiled, not run on hardware
//
#include <hip/hip_runtime.h>
#include <hip/hip_bf16.h>

// ---------------------------------------------------------------------------
// MixLoRA sparse-MoE forward for MI455X (gfx1250, wave32, WMMA bf16).
//   final = sum_e w[t,e] * ( silu(x@Wg_eff + bg) * (x@Wu_eff + bu) @ Wd_eff + bd )
// LoRA folded exactly into effective weights (low-rank weight delta),
// materialized PRE-TRANSPOSED (n-major, K contiguous) so GEMM tile staging is
// a pure 2D copy via GLOBAL_LOAD_ASYNC_TO_LDS_B128, double-buffered so the
// async engine overlaps with WMMA compute (ASYNCcnt software pipeline).
// ---------------------------------------------------------------------------

typedef __bf16 bf16;
typedef __attribute__((ext_vector_type(16))) __bf16 v16bf;
typedef __attribute__((ext_vector_type(8)))  float  v8f;

constexpr int   kT = 4096, kD = 1024, kFF = 2048, kE = 4, kR = 16;
constexpr float kLoraScale = 2.0f;   // alpha / r = 32/16

#define DEV_INLINE static __device__ __forceinline__

DEV_INLINE bf16 f2b(float f) {                 // RNE float->bf16
  unsigned u = __float_as_uint(f);
  u += 0x7fffu + ((u >> 16) & 1u);
  unsigned short h = (unsigned short)(u >> 16);
  return *reinterpret_cast<bf16*>(&h);
}

union Frag16 { v16bf v; uint4 q[2]; };

// CDNA5 async copy: 16 bytes global -> LDS per lane, tracked by ASYNCcnt.
// Generic LDS pointer low 32 bits == wave-relative LDS byte offset.
DEV_INLINE void async_ld16(const bf16* g, bf16* l) {
  unsigned loff = (unsigned)(unsigned long long)(const void*)l;
  asm volatile("global_load_async_to_lds_b128 %0, %1, off"
               :: "v"(loff), "v"(g) : "memory");
}
template <int N> DEV_INLINE void wait_async_le() {
  asm volatile("s_wait_asynccnt %0" :: "i"(N) : "memory");
}

// ---------------------------------------------------------------------------
// Kernel 1: router.  One wave per token: logits = x . gate_w[e], softmax,
// top-2, renormalize.  Writes dense combine weights [T,E] and logits output.
// ---------------------------------------------------------------------------
__global__ __launch_bounds__(256) void router_topk(
    const float* __restrict__ x, const float* __restrict__ gw,
    float* __restrict__ wcomb, float* __restrict__ logits_out) {
  const int t    = blockIdx.x * 8 + (threadIdx.x >> 5);
  const int lane = threadIdx.x & 31;
  float a0 = 0.f, a1 = 0.f, a2 = 0.f, a3 = 0.f;
  for (int d = lane; d < kD; d += 32) {
    float xv = x[(size_t)t * kD + d];
    a0 += xv * gw[0 * kD + d];
    a1 += xv * gw[1 * kD + d];
    a2 += xv * gw[2 * kD + d];
    a3 += xv * gw[3 * kD + d];
  }
#pragma unroll
  for (int off = 16; off > 0; off >>= 1) {
    a0 += __shfl_down(a0, off, 32);
    a1 += __shfl_down(a1, off, 32);
    a2 += __shfl_down(a2, off, 32);
    a3 += __shfl_down(a3, off, 32);
  }
  if (lane == 0) {
    float l[4] = {a0, a1, a2, a3};
    int i1 = 0;
#pragma unroll
    for (int e = 1; e < 4; ++e) if (l[e] > l[i1]) i1 = e;
    int i2 = (i1 == 0) ? 1 : 0;
#pragma unroll
    for (int e = 0; e < 4; ++e) if (e != i1 && l[e] > l[i2]) i2 = e;
    float m  = fmaxf(l[i1], l[i2]);
    float e1 = __expf(l[i1] - m), e2 = __expf(l[i2] - m);
    float inv = 1.0f / (e1 + e2);
#pragma unroll
    for (int e = 0; e < 4; ++e) {
      float wv = (e == i1) ? e1 * inv : (e == i2) ? e2 * inv : 0.0f;
      wcomb[(size_t)t * kE + e]      = wv;
      logits_out[(size_t)t * kE + e] = l[e];
    }
  }
}

// ---------------------------------------------------------------------------
// Kernel 2: convert activations to bf16.
// ---------------------------------------------------------------------------
__global__ __launch_bounds__(256) void cvt_x_bf16(const float* __restrict__ x,
                                                  bf16* __restrict__ xb) {
  size_t i = ((size_t)blockIdx.x * 256 + threadIdx.x) * 8;
#pragma unroll
  for (int j = 0; j < 8; ++j) xb[i + j] = f2b(x[i + j]);
}

// ---------------------------------------------------------------------------
// Kernel 3: effective gate/up weight, TRANSPOSED to [E, 2FF, D] bf16.
//   WguT[e,f,d] = gate_up_w[e,d,f] + scale * sum_r A[e,r,d] * B[e,f,r]
// 32x32 tile transpose through padded LDS: coalesced reads AND writes.
// ---------------------------------------------------------------------------
__global__ __launch_bounds__(256) void build_wgu_eff_t(
    const float* __restrict__ guw,
    const float* __restrict__ Ag, const float* __restrict__ Bg,
    const float* __restrict__ Au, const float* __restrict__ Bu,
    bf16* __restrict__ outT) {
  __shared__ float tile[32][33];
  const int e   = blockIdx.z;
  const int f0  = blockIdx.x * 32;   // over [0, 2FF)
  const int d0  = blockIdx.y * 32;   // over [0, D)
  const int tid = threadIdx.x;
  {
    int fi = tid & 31, dq = tid >> 5;
    int f  = f0 + fi;
    const float* A;
    const float* B;
    if (f < kFF) { A = Ag + (size_t)e * kR * kD; B = Bg + ((size_t)e * kFF + f) * kR; }
    else         { A = Au + (size_t)e * kR * kD; B = Bu + ((size_t)e * kFF + (f - kFF)) * kR; }
#pragma unroll
    for (int j = 0; j < 4; ++j) {
      int d = d0 + dq * 4 + j;
      float lora = 0.f;
#pragma unroll
      for (int r = 0; r < kR; ++r) lora += A[(size_t)r * kD + d] * B[r];
      float base = guw[((size_t)e * kD + d) * (2 * kFF) + f];
      tile[fi][dq * 4 + j] = base + kLoraScale * lora;
    }
  }
  __syncthreads();
  {
    int di = tid & 31, fq = tid >> 5;
#pragma unroll
    for (int j = 0; j < 4; ++j) {
      int fi = fq * 4 + j;
      outT[((size_t)e * (2 * kFF) + f0 + fi) * kD + d0 + di] = f2b(tile[fi][di]);
    }
  }
}

// ---------------------------------------------------------------------------
// Kernel 4: effective down weight, TRANSPOSED to [E, D, FF] bf16.
//   WdT[e,d,f] = down_w[e,f,d] + scale * sum_r A_down[e,r,f] * B_down[e,d,r]
// ---------------------------------------------------------------------------
__global__ __launch_bounds__(256) void build_wd_eff_t(
    const float* __restrict__ dw,
    const float* __restrict__ Ad, const float* __restrict__ Bd,
    bf16* __restrict__ outT) {
  __shared__ float tile[32][33];
  const int e   = blockIdx.z;
  const int d0  = blockIdx.x * 32;   // over [0, D)
  const int f0  = blockIdx.y * 32;   // over [0, FF)
  const int tid = threadIdx.x;
  {
    int di = tid & 31, fq = tid >> 5;
    int d  = d0 + di;
    const float* B = Bd + ((size_t)e * kD + d) * kR;
#pragma unroll
    for (int j = 0; j < 4; ++j) {
      int f = f0 + fq * 4 + j;
      float lora = 0.f;
#pragma unroll
      for (int r = 0; r < kR; ++r) lora += Ad[((size_t)e * kR + r) * kFF + f] * B[r];
      float base = dw[((size_t)e * kFF + f) * kD + d];
      tile[di][fq * 4 + j] = base + kLoraScale * lora;
    }
  }
  __syncthreads();
  {
    int fi = tid & 31, dq = tid >> 5;
#pragma unroll
    for (int j = 0; j < 4; ++j) {
      int di = dq * 4 + j;
      outT[((size_t)e * kD + d0 + di) * kFF + f0 + fi] = f2b(tile[di][fi]);
    }
  }
}

// ---------------------------------------------------------------------------
// Kernel 5: GEMM1 + SwiGLU.  act[e] = silu(xb@Wg_eff+bg) * (xb@Wu_eff+bu).
// 256 threads / 8 waves; BM=128 x BN=64 gate cols (+ paired up cols); wave
// tile 32x32; K-chunks of 32.  Double-buffered async global->LDS pipeline:
// tile k+1 streams in while tile k is consumed by WMMA.
// ---------------------------------------------------------------------------
__global__ __launch_bounds__(256) void gemm1_swiglu(
    const bf16* __restrict__ xb,       // [T,D]
    const bf16* __restrict__ wguT,     // [E,2FF,D] bf16 effective, transposed
    const float* __restrict__ gub,     // [E,2FF]
    bf16* __restrict__ act) {          // [E,T,FF]
  __shared__ alignas(16) bf16 lds_a [2][128 * 32];   // [m][k]
  __shared__ alignas(16) bf16 lds_bg[2][ 64 * 32];   // [n][k]
  __shared__ alignas(16) bf16 lds_bu[2][ 64 * 32];   // [n][k]

  const int e    = blockIdx.z;
  const int n0   = blockIdx.x * 64;    // gate-column base in [0,FF)
  const int m0   = blockIdx.y * 128;   // token-row base
  const int tid  = threadIdx.x;
  const int lane = tid & 31;
  const int wid  = tid >> 5;
  const int wm   = (wid & 3) * 32;
  const int wn   = (wid >> 2) * 32;

  const bf16* wgt = wguT + (size_t)e * (2 * kFF) * kD;

  v8f accg[2][2] = {};
  v8f accu[2][2] = {};

  const int kb_a = (lane < 16) ? 0 : 8;
  const int kb_b = (lane < 16) ? 0 : 8;

  // stage one K-chunk (4 async b128 per thread) into buffer b
  auto stage = [&](int k0, int b) {
#pragma unroll
    for (int c = 0; c < 2; ++c) {
      int chunk = tid + c * 256;            // 512 chunks of 8 bf16
      int m  = chunk >> 2;
      int ko = (chunk & 3) * 8;
      async_ld16(xb + (size_t)(m0 + m) * kD + k0 + ko, &lds_a[b][m * 32 + ko]);
    }
    int n  = tid >> 2;                      // 0..63
    int ko = (tid & 3) * 8;                 // 0,8,16,24
    async_ld16(wgt + (size_t)(n0 + n) * kD + k0 + ko,       &lds_bg[b][n * 32 + ko]);
    async_ld16(wgt + (size_t)(kFF + n0 + n) * kD + k0 + ko, &lds_bu[b][n * 32 + ko]);
  };

  constexpr int nK = kD / 32;
  stage(0, 0);
  for (int kt = 0; kt < nK; ++kt) {
    const int cur = kt & 1;
    if (kt + 1 < nK) {            // prefetch next tile, wait only for current
      stage((kt + 1) * 32, cur ^ 1);
      wait_async_le<4>();
    } else {
      wait_async_le<0>();
    }
    __syncthreads();

    Frag16 a[2], bg[2], bu[2];
#pragma unroll
    for (int mi = 0; mi < 2; ++mi) {
      int row = wm + mi * 16 + (lane & 15);
      a[mi].q[0] = *(const uint4*)&lds_a[cur][row * 32 + kb_a];
      a[mi].q[1] = *(const uint4*)&lds_a[cur][row * 32 + kb_a + 16];
    }
#pragma unroll
    for (int ni = 0; ni < 2; ++ni) {
      int col = wn + ni * 16 + (lane & 15);
      bg[ni].q[0] = *(const uint4*)&lds_bg[cur][col * 32 + kb_b];
      bg[ni].q[1] = *(const uint4*)&lds_bg[cur][col * 32 + kb_b + 16];
      bu[ni].q[0] = *(const uint4*)&lds_bu[cur][col * 32 + kb_b];
      bu[ni].q[1] = *(const uint4*)&lds_bu[cur][col * 32 + kb_b + 16];
    }
#pragma unroll
    for (int mi = 0; mi < 2; ++mi)
#pragma unroll
      for (int ni = 0; ni < 2; ++ni) {
        accg[mi][ni] = __builtin_amdgcn_wmma_f32_16x16x32_bf16(
            false, a[mi].v, false, bg[ni].v, (short)0, accg[mi][ni], false, false);
        accu[mi][ni] = __builtin_amdgcn_wmma_f32_16x16x32_bf16(
            false, a[mi].v, false, bu[ni].v, (short)0, accu[mi][ni], false, false);
      }
    __syncthreads();   // protect buffer 'cur' before it is re-staged
  }

  // ---- epilogue: bias, SiLU(gate)*up, bf16 store -------------------------
  const int rofs = (lane < 16) ? 0 : 8;
#pragma unroll
  for (int mi = 0; mi < 2; ++mi) {
#pragma unroll
    for (int ni = 0; ni < 2; ++ni) {
      int col   = n0 + wn + ni * 16 + (lane & 15);
      float bgs = gub[(size_t)e * (2 * kFF) + col];
      float bus = gub[(size_t)e * (2 * kFF) + kFF + col];
#pragma unroll
      for (int r = 0; r < 8; ++r) {
        int row = m0 + wm + mi * 16 + rofs + r;
        float g = accg[mi][ni][r] + bgs;
        float u = accu[mi][ni][r] + bus;
        float s = g / (1.0f + __expf(-g));
        act[((size_t)e * kT + row) * kFF + col] = f2b(s * u);
      }
    }
  }
}

// ---------------------------------------------------------------------------
// Kernel 6: GEMM2 + weighted expert combine.
//   final[t,d] = sum_e w[t,e] * (act[e] @ Wd_eff[e] + down_b[e])[t,d]
// Expert loop in-register; same double-buffered async pipeline over K=FF.
// ---------------------------------------------------------------------------
__global__ __launch_bounds__(256) void gemm2_combine(
    const bf16* __restrict__ act,      // [E,T,FF]
    const bf16* __restrict__ wdT,      // [E,D,FF] bf16 effective, transposed
    const float* __restrict__ db,      // [E,D]
    const float* __restrict__ wcomb,   // [T,E]
    float* __restrict__ out) {         // [T,D]
  __shared__ alignas(16) bf16 lds_a[2][128 * 32];
  __shared__ alignas(16) bf16 lds_b[2][ 64 * 32];

  const int n0   = blockIdx.x * 64;    // D-column base
  const int m0   = blockIdx.y * 128;   // token-row base
  const int tid  = threadIdx.x;
  const int lane = tid & 31;
  const int wid  = tid >> 5;
  const int wm   = (wid & 3) * 32;
  const int wn   = (wid >> 2) * 32;

  const int kb_a = (lane < 16) ? 0 : 8;
  const int kb_b = (lane < 16) ? 0 : 8;
  const int rofs = (lane < 16) ? 0 : 8;

  v8f fin[2][2] = {};

  for (int e = 0; e < kE; ++e) {
    const bf16* ae = act + (size_t)e * kT * kFF;
    const bf16* we = wdT + (size_t)e * kD * kFF;
    v8f part[2][2] = {};

    auto stage = [&](int k0, int b) {
#pragma unroll
      for (int c = 0; c < 2; ++c) {
        int chunk = tid + c * 256;
        int m  = chunk >> 2;
        int ko = (chunk & 3) * 8;
        async_ld16(ae + (size_t)(m0 + m) * kFF + k0 + ko, &lds_a[b][m * 32 + ko]);
      }
      int n  = tid >> 2;
      int ko = (tid & 3) * 8;
      async_ld16(we + (size_t)(n0 + n) * kFF + k0 + ko, &lds_b[b][n * 32 + ko]);
    };

    constexpr int nK = kFF / 32;
    stage(0, 0);
    for (int kt = 0; kt < nK; ++kt) {
      const int cur = kt & 1;
      if (kt + 1 < nK) {
        stage((kt + 1) * 32, cur ^ 1);
        wait_async_le<3>();
      } else {
        wait_async_le<0>();
      }
      __syncthreads();

      Frag16 a[2], b[2];
#pragma unroll
      for (int mi = 0; mi < 2; ++mi) {
        int row = wm + mi * 16 + (lane & 15);
        a[mi].q[0] = *(const uint4*)&lds_a[cur][row * 32 + kb_a];
        a[mi].q[1] = *(const uint4*)&lds_a[cur][row * 32 + kb_a + 16];
      }
#pragma unroll
      for (int ni = 0; ni < 2; ++ni) {
        int col = wn + ni * 16 + (lane & 15);
        b[ni].q[0] = *(const uint4*)&lds_b[cur][col * 32 + kb_b];
        b[ni].q[1] = *(const uint4*)&lds_b[cur][col * 32 + kb_b + 16];
      }
#pragma unroll
      for (int mi = 0; mi < 2; ++mi)
#pragma unroll
        for (int ni = 0; ni < 2; ++ni)
          part[mi][ni] = __builtin_amdgcn_wmma_f32_16x16x32_bf16(
              false, a[mi].v, false, b[ni].v, (short)0, part[mi][ni], false, false);
      __syncthreads();
    }

    // fin += w[row,e] * (part + down_b[e,col])
#pragma unroll
    for (int mi = 0; mi < 2; ++mi) {
      float wv8[8];
#pragma unroll
      for (int r = 0; r < 8; ++r) {
        int row = m0 + wm + mi * 16 + rofs + r;
        wv8[r] = wcomb[(size_t)row * kE + e];
      }
#pragma unroll
      for (int ni = 0; ni < 2; ++ni) {
        int col  = n0 + wn + ni * 16 + (lane & 15);
        float bb = db[(size_t)e * kD + col];
#pragma unroll
        for (int r = 0; r < 8; ++r)
          fin[mi][ni][r] += wv8[r] * (part[mi][ni][r] + bb);
      }
    }
  }

#pragma unroll
  for (int mi = 0; mi < 2; ++mi)
#pragma unroll
    for (int ni = 0; ni < 2; ++ni) {
      int col = n0 + wn + ni * 16 + (lane & 15);
#pragma unroll
      for (int r = 0; r < 8; ++r) {
        int row = m0 + wm + mi * 16 + rofs + r;
        out[(size_t)row * kD + col] = fin[mi][ni][r];
      }
    }
}

// ---------------------------------------------------------------------------
// Launch.  Workspace (~126 MB):
//   wcomb[T*E] f32 | xb[T*D] bf16 | WguT[E*2FF*D] bf16 | WdT[E*D*FF] bf16 |
//   act[E*T*FF] bf16
// ---------------------------------------------------------------------------
extern "C" void kernel_launch(void* const* d_in, const int* in_sizes, int n_in,
                              void* d_out, int out_size, void* d_ws, size_t ws_size,
                              hipStream_t stream) {
  const float* x      = (const float*)d_in[0];
  const float* gate_w = (const float*)d_in[1];
  const float* guw    = (const float*)d_in[2];
  const float* gub    = (const float*)d_in[3];
  const float* dww    = (const float*)d_in[4];
  const float* dwb    = (const float*)d_in[5];
  const float* Ag     = (const float*)d_in[6];
  const float* Bg     = (const float*)d_in[7];
  const float* Au     = (const float*)d_in[8];
  const float* Bu     = (const float*)d_in[9];
  const float* Ad     = (const float*)d_in[10];
  const float* Bd     = (const float*)d_in[11];

  float* out        = (float*)d_out;
  float* logits_out = out + (size_t)kT * kD;

  char* ws = (char*)d_ws;
  size_t off = 0;
  float* wcomb = (float*)(ws + off); off += 64 * 1024;
  bf16*  xb    = (bf16*) (ws + off); off += (size_t)kT * kD * 2;
  bf16*  wguT  = (bf16*) (ws + off); off += (size_t)kE * (2 * kFF) * kD * 2;
  bf16*  wdT   = (bf16*) (ws + off); off += (size_t)kE * kD * kFF * 2;
  bf16*  actb  = (bf16*) (ws + off); off += (size_t)kE * kT * kFF * 2;

  router_topk<<<kT / 8, 256, 0, stream>>>(x, gate_w, wcomb, logits_out);
  cvt_x_bf16<<<(int)((size_t)kT * kD / 2048), 256, 0, stream>>>(x, xb);

  dim3 gt1(2 * kFF / 32, kD / 32, kE);     // 128 x 32 x 4
  build_wgu_eff_t<<<gt1, 256, 0, stream>>>(guw, Ag, Bg, Au, Bu, wguT);
  dim3 gt2(kD / 32, kFF / 32, kE);         // 32 x 64 x 4
  build_wd_eff_t<<<gt2, 256, 0, stream>>>(dww, Ad, Bd, wdT);

  dim3 g1(kFF / 64, kT / 128, kE);
  gemm1_swiglu<<<g1, 256, 0, stream>>>(xb, wguT, gub, actb);

  dim3 g2(kD / 64, kT / 128, 1);
  gemm2_combine<<<g2, 256, 0, stream>>>(actb, wdT, dwb, wcomb, out);
}
